// NEQUIPLayerFlax_68676527063644
// MI455X (gfx1250) — matrix operand order repaired and checked by reading, not compiled, not run on hardware
//
#include <hip/hip_runtime.h>
#include <hip/hip_fp16.h>

// ---------------------------------------------------------------------------
// CDNA5 (gfx1250) NequIP layer, mixed precision:
//   dense GEMMs -> v_wmma_f32_16x16x32_f16 (f16 in, f32 accumulate)
//   gather/scatter -> L2-resident global_atomic_add_f32
// Tile bodies are templated on FULL so the (ubiquitous) full-tile path is
// completely branchless; tail tiles take a separate cold instantiation.
// ---------------------------------------------------------------------------

typedef __attribute__((ext_vector_type(16))) _Float16 v16h;
typedef __attribute__((ext_vector_type(8)))  _Float16 v8h;
typedef __attribute__((ext_vector_type(8)))  float    v8f;

#define SQRT3F      1.7320508075688772f
#define INV_SQRT3F  0.5773502691896258f
#define INV_SQRT8F  0.35355339059327373f
#define INV_2M_F    0.08838834764831845f   /* 1/sqrt(128) */

__device__ __forceinline__ float silu_f(float x) { return x / (1.0f + __expf(-x)); }

__device__ __forceinline__ v8f wmma32(v16h a, v16h b, v8f c) {
  // D = A(16x32 f16) * B(32x16 f16) + C(16x16 f32)
  return __builtin_amdgcn_wmma_f32_16x16x32_f16(false, a, false, b, (short)0, c,
                                                false, false);
}

// A fragment (16x32, MxK) from LDS row-major [16][ldk] f16.
// Lane L (row M=L&15): halves 0..7  = K = kbase + (L<16?0:8)  + i
//                      halves 8..15 = K = kbase + 16 + (L<16?0:8) + i
__device__ __forceinline__ v16h load_A_frag(const _Float16* act, int ldk,
                                            int kbase, int lane) {
  int m = lane & 15;
  int o = (lane < 16) ? 0 : 8;
  const _Float16* p = act + m * ldk + kbase + o;
  v8h lo = *(const v8h*)(p);
  v8h hi = *(const v8h*)(p + 16);
  v16h a;
#pragma unroll
  for (int i = 0; i < 8; ++i) { a[i] = lo[i]; a[i + 8] = hi[i]; }
  return a;
}

// B fragment (32x16, KxN) from LDS transposed weight [Nfull][ldk] f16.
// Lane L (col N=nbase+(L&15)): halves 0..7  = K = kbase + (L<16?0:8)  + i
//                             halves 8..15 = K = kbase + 16 + (L<16?0:8) + i
__device__ __forceinline__ v16h load_B_frag(const _Float16* wt, int ldk,
                                            int nbase, int kbase, int lane) {
  int n = nbase + (lane & 15);
  int o = (lane < 16) ? 0 : 8;
  const _Float16* p = wt + n * ldk + kbase + o;
  v8h lo = *(const v8h*)(p);
  v8h hi = *(const v8h*)(p + 16);
  v16h b;
#pragma unroll
  for (int i = 0; i < 8; ++i) { b[i] = lo[i]; b[i + 8] = hi[i]; }
  return b;
}

// Store D fragment (C/D layout: VGPR r -> row M=r+(lane<16?0:8), col N=lane&15)
// to LDS activation buffer with fused scale + SiLU, converted to f16.
__device__ __forceinline__ void store_act_silu(const v8f& acc, _Float16* act,
                                               int ldk, int nbase, int lane,
                                               float scale) {
  int n  = nbase + (lane & 15);
  int mo = (lane < 16) ? 0 : 8;
#pragma unroll
  for (int r = 0; r < 8; ++r) {
    float x = acc[r] * scale;
    x = silu_f(x);
    act[(r + mo) * ldk + n] = (_Float16)x;
  }
}

// ---------------------------------------------------------------------------
__global__ void k_zero(float* __restrict__ p, long n) {
  long t = blockIdx.x * (long)blockDim.x + threadIdx.x;
  long stride = (long)gridDim.x * blockDim.x;
  for (; t < n; t += stride) p[t] = 0.0f;
}

// ---------------------------------------------------------------------------
// up_s = s @ W_up_s / 8 ; up_v[:,:,c] = v_c @ W_up_v / 8
template <bool FULL>
__device__ __forceinline__ void up_tile(const float* __restrict__ node_feats,
                                        const _Float16* wus, const _Float16* wuv,
                                        _Float16* act,
                                        float* __restrict__ up_s,
                                        float* __restrict__ up_v,
                                        int base, int n_nodes, int lane) {
  const int nloc = lane & 15, mo = (lane < 16) ? 0 : 8;
  // ---- up_s ----
  for (int i = lane; i < 1024; i += 32) {
    int m = i >> 6, k = i & 63;
    int node = base + m;
    if (!FULL && node >= n_nodes) node = n_nodes - 1;
    act[i] = (_Float16)node_feats[node * 256 + k];
  }
#pragma unroll
  for (int nt = 0; nt < 4; ++nt) {
    v8f acc = {};
#pragma unroll
    for (int kt = 0; kt < 2; ++kt) {
      v16h a = load_A_frag(act, 64, kt * 32, lane);
      v16h b = load_B_frag(wus, 64, nt * 16, kt * 32, lane);
      acc = wmma32(a, b, acc);
    }
    int n = nt * 16 + nloc;
#pragma unroll
    for (int r = 0; r < 8; ++r) {
      int row = base + r + mo;
      if (FULL || row < n_nodes) up_s[row * 64 + n] = acc[r] * 0.125f;
    }
  }
  // ---- up_v, 3 components ----
  for (int c = 0; c < 3; ++c) {
    for (int i = lane; i < 1024; i += 32) {
      int m = i >> 6, k = i & 63;
      int node = base + m;
      if (!FULL && node >= n_nodes) node = n_nodes - 1;
      act[i] = (_Float16)node_feats[node * 256 + 64 + k * 3 + c];
    }
#pragma unroll
    for (int nt = 0; nt < 4; ++nt) {
      v8f acc = {};
#pragma unroll
      for (int kt = 0; kt < 2; ++kt) {
        v16h a = load_A_frag(act, 64, kt * 32, lane);
        v16h b = load_B_frag(wuv, 64, nt * 16, kt * 32, lane);
        acc = wmma32(a, b, acc);
      }
      int n = nt * 16 + nloc;
#pragma unroll
      for (int r = 0; r < 8; ++r) {
        int row = base + r + mo;
        if (FULL || row < n_nodes) up_v[(row * 64 + n) * 3 + c] = acc[r] * 0.125f;
      }
    }
  }
}

__global__ void k_up(const float* __restrict__ node_feats,
                     const float* __restrict__ W_up_s,
                     const float* __restrict__ W_up_v,
                     float* __restrict__ up_s, float* __restrict__ up_v,
                     int n_nodes) {
  extern __shared__ char smem[];
  _Float16* wus = (_Float16*)smem;       // [64][64] transposed (N,K)
  _Float16* wuv = wus + 4096;            // [64][64] transposed
  const int tid = threadIdx.x;
  for (int i = tid; i < 4096; i += blockDim.x) {
    int n = i >> 6, k = i & 63;
    wus[i] = (_Float16)W_up_s[k * 64 + n];
    wuv[i] = (_Float16)W_up_v[k * 64 + n];
  }
  __syncthreads();
  const int lane = tid & 31, wave = tid >> 5;
  _Float16* act = wuv + 4096 + wave * 1024;   // [16][64]
  int n_tiles = (n_nodes + 15) >> 4;
  int tile = blockIdx.x * (blockDim.x >> 5) + wave;
  if (tile >= n_tiles) return;
  int base = tile * 16;
  if (base + 16 <= n_nodes)
    up_tile<true>(node_feats, wus, wuv, act, up_s, up_v, base, n_nodes, lane);
  else
    up_tile<false>(node_feats, wus, wuv, act, up_s, up_v, base, n_nodes, lane);
}

// ---------------------------------------------------------------------------
// Edge tile: radial MLP (4 GEMMs, all WMMA) fused with tensor-product
// message formation and atomic scatter-add into agg_s/agg_v.
// The 16 mix fragments are split into 4 groups so the message-type branch is
// fully uniform (no per-lane divergence).
template <bool FULL>
__device__ __forceinline__ void edge_tile(
    const float* __restrict__ vectors, const float* __restrict__ radial,
    const int* __restrict__ senders, const int* __restrict__ receivers,
    const _Float16* w1t, const _Float16* w2t, const _Float16* w3t,
    const _Float16* wot, _Float16* actA, _Float16* actB,
    float* ymeta, int* sr,
    const float* __restrict__ up_s, const float* __restrict__ up_v,
    float* __restrict__ agg_s, float* __restrict__ agg_v,
    int ebase, int n_edges, int lane) {
  // per-edge geometry + indices
  if (lane < 16) {
    int e = ebase + lane;
    float vx = 0, vy = 0, vz = 0; int snd = 0, rcv = 0;
    if (FULL || e < n_edges) {
      vx = vectors[e * 3]; vy = vectors[e * 3 + 1]; vz = vectors[e * 3 + 2];
      snd = senders[e]; rcv = receivers[e];
    }
    float len = sqrtf(vx * vx + vy * vy + vz * vz);
    float valid = ((FULL || e < n_edges) && len != 0.0f) ? 1.0f : 0.0f;
    float inv = (len != 0.0f) ? (SQRT3F / len) : 0.0f;
    ymeta[lane * 4 + 0] = vx * inv;
    ymeta[lane * 4 + 1] = vy * inv;
    ymeta[lane * 4 + 2] = vz * inv;
    ymeta[lane * 4 + 3] = valid;
    sr[lane * 2] = snd; sr[lane * 2 + 1] = rcv;
  }
  // radial embedding -> actA (K padded to 32 with zeros)
  for (int i = lane; i < 1024; i += 32) actA[i] = (_Float16)0.0f;
  for (int i = lane; i < 128; i += 32) {
    int m = i >> 3, k = i & 7;
    int e = ebase + m;
    actA[m * 64 + k] = (FULL || e < n_edges) ? (_Float16)radial[e * 8 + k]
                                             : (_Float16)0.0f;
  }
  // h1 = silu(r @ w1 / sqrt(8)) -> actB
  {
    v16h a = load_A_frag(actA, 64, 0, lane);
#pragma unroll
    for (int nt = 0; nt < 4; ++nt) {
      v16h b = load_B_frag(w1t, 32, nt * 16, 0, lane);
      v8f acc = {};
      acc = wmma32(a, b, acc);
      store_act_silu(acc, actB, 64, nt * 16, lane, INV_SQRT8F);
    }
  }
  // h2 = silu(h1 @ w2 / 8) -> actA
#pragma unroll
  for (int nt = 0; nt < 4; ++nt) {
    v8f acc = {};
#pragma unroll
    for (int kt = 0; kt < 2; ++kt) {
      v16h a = load_A_frag(actB, 64, kt * 32, lane);
      v16h b = load_B_frag(w2t, 64, nt * 16, kt * 32, lane);
      acc = wmma32(a, b, acc);
    }
    store_act_silu(acc, actA, 64, nt * 16, lane, 0.125f);
  }
  // h3 = silu(h2 @ w3 / 8) -> actB
#pragma unroll
  for (int nt = 0; nt < 4; ++nt) {
    v8f acc = {};
#pragma unroll
    for (int kt = 0; kt < 2; ++kt) {
      v16h a = load_A_frag(actA, 64, kt * 32, lane);
      v16h b = load_B_frag(w3t, 64, nt * 16, kt * 32, lane);
      acc = wmma32(a, b, acc);
    }
    store_act_silu(acc, actB, 64, nt * 16, lane, 0.125f);
  }

  // Hoist this lane's 8 edge rows (geometry + indices) into registers.
  const int nloc = lane & 15, mo = (lane < 16) ? 0 : 8;
  float ey0[8], ey1[8], ey2[8], evl[8];
  int esnd[8], ercv[8];
#pragma unroll
  for (int r = 0; r < 8; ++r) {
    int me = r + mo;
    ey0[r] = ymeta[me * 4 + 0];
    ey1[r] = ymeta[me * 4 + 1];
    ey2[r] = ymeta[me * 4 + 2];
    evl[r] = ymeta[me * 4 + 3] * 0.125f;     // fold mix scale into validity
    esnd[r] = sr[me * 2];
    ercv[r] = sr[me * 2 + 1];
  }

  // mix = h3 @ wo / 8 ; fragments consumed in C/D layout, scatter messages.
  // Group 1 (nt 0..3, chan 0..63): msg_s = es * mix
#pragma unroll
  for (int nt = 0; nt < 4; ++nt) {
    v8f acc = {};
#pragma unroll
    for (int kt = 0; kt < 2; ++kt) {
      v16h a = load_A_frag(actB, 64, kt * 32, lane);
      v16h b = load_B_frag(wot, 64, nt * 16, kt * 32, lane);
      acc = wmma32(a, b, acc);
    }
    int chan = nt * 16 + nloc;
#pragma unroll
    for (int r = 0; r < 8; ++r) {
      float mval = acc[r] * evl[r];
      float esv = up_s[esnd[r] * 64 + chan];
      atomicAdd(&agg_s[ercv[r] * 128 + chan], esv * mval);
    }
  }
  // Group 2 (nt 4..7, chan 64..127): msg_s = tp_s * mix
#pragma unroll
  for (int nt = 4; nt < 8; ++nt) {
    v8f acc = {};
#pragma unroll
    for (int kt = 0; kt < 2; ++kt) {
      v16h a = load_A_frag(actB, 64, kt * 32, lane);
      v16h b = load_B_frag(wot, 64, nt * 16, kt * 32, lane);
      acc = wmma32(a, b, acc);
    }
    int chan = nt * 16 + nloc;
    int k = chan - 64;
#pragma unroll
    for (int r = 0; r < 8; ++r) {
      float mval = acc[r] * evl[r];
      const float* ev = up_v + (esnd[r] * 64 + k) * 3;
      float tps = (ev[0] * ey0[r] + ev[1] * ey1[r] + ev[2] * ey2[r]) * INV_SQRT3F;
      atomicAdd(&agg_s[ercv[r] * 128 + chan], tps * mval);
    }
  }
  // Group 3 (nt 8..11, chan 128..191): msg_v = ev * mix
#pragma unroll
  for (int nt = 8; nt < 12; ++nt) {
    v8f acc = {};
#pragma unroll
    for (int kt = 0; kt < 2; ++kt) {
      v16h a = load_A_frag(actB, 64, kt * 32, lane);
      v16h b = load_B_frag(wot, 64, nt * 16, kt * 32, lane);
      acc = wmma32(a, b, acc);
    }
    int k = nt * 16 + nloc - 128;
#pragma unroll
    for (int r = 0; r < 8; ++r) {
      float mval = acc[r] * evl[r];
      const float* ev = up_v + (esnd[r] * 64 + k) * 3;
      float* dst = agg_v + (ercv[r] * 128 + k) * 3;
      atomicAdd(dst + 0, ev[0] * mval);
      atomicAdd(dst + 1, ev[1] * mval);
      atomicAdd(dst + 2, ev[2] * mval);
    }
  }
  // Group 4 (nt 12..15, chan 192..255): msg_v = (es (x) y) * mix
#pragma unroll
  for (int nt = 12; nt < 16; ++nt) {
    v8f acc = {};
#pragma unroll
    for (int kt = 0; kt < 2; ++kt) {
      v16h a = load_A_frag(actB, 64, kt * 32, lane);
      v16h b = load_B_frag(wot, 64, nt * 16, kt * 32, lane);
      acc = wmma32(a, b, acc);
    }
    int k = nt * 16 + nloc - 192;
#pragma unroll
    for (int r = 0; r < 8; ++r) {
      float esv = up_s[esnd[r] * 64 + k] * acc[r] * evl[r];
      float* dst = agg_v + (ercv[r] * 128 + 64 + k) * 3;
      atomicAdd(dst + 0, esv * ey0[r]);
      atomicAdd(dst + 1, esv * ey1[r]);
      atomicAdd(dst + 2, esv * ey2[r]);
    }
  }
}

__global__ void k_edge(const float* __restrict__ vectors,
                       const float* __restrict__ radial,
                       const int* __restrict__ senders,
                       const int* __restrict__ receivers,
                       const float* __restrict__ w1, const float* __restrict__ w2,
                       const float* __restrict__ w3, const float* __restrict__ wo,
                       const float* __restrict__ up_s, const float* __restrict__ up_v,
                       float* __restrict__ agg_s, float* __restrict__ agg_v,
                       int n_edges) {
  extern __shared__ char smem[];
  _Float16* w1t   = (_Float16*)smem;       // [64][32]  (K padded 8->32)
  _Float16* w2t   = w1t + 2048;            // [64][64]
  _Float16* w3t   = w2t + 4096;            // [64][64]
  _Float16* wot   = w3t + 4096;            // [256][64]
  _Float16* acts0 = wot + 16384;           // 2 waves * (actA+actB = 2048 halves)
  float*    metaf = (float*)(acts0 + 4096);
  const int tid = threadIdx.x;
  for (int i = tid; i < 2048; i += blockDim.x) {
    int n = i >> 5, k = i & 31;
    w1t[i] = (k < 8) ? (_Float16)w1[k * 64 + n] : (_Float16)0.0f;
  }
  for (int i = tid; i < 4096; i += blockDim.x) {
    int n = i >> 6, k = i & 63;
    w2t[i] = (_Float16)w2[k * 64 + n];
    w3t[i] = (_Float16)w3[k * 64 + n];
  }
  for (int i = tid; i < 16384; i += blockDim.x) {
    int n = i >> 6, k = i & 63;
    wot[i] = (_Float16)wo[k * 256 + n];
  }
  __syncthreads();
  const int lane = tid & 31, wave = tid >> 5;
  _Float16* actA = acts0 + wave * 2048;    // [16][64]
  _Float16* actB = actA + 1024;            // [16][64]
  float* ymeta = metaf + wave * 96;        // [16][4]: y*sqrt3 (3) + valid
  int*   sr    = (int*)(ymeta + 64);       // [16][2]: sender, receiver
  int n_tiles = (n_edges + 15) >> 4;
  int tile = blockIdx.x * (blockDim.x >> 5) + wave;
  if (tile >= n_tiles) return;
  int ebase = tile * 16;
  if (ebase + 16 <= n_edges)
    edge_tile<true>(vectors, radial, senders, receivers, w1t, w2t, w3t, wot,
                    actA, actB, ymeta, sr, up_s, up_v, agg_s, agg_v,
                    ebase, n_edges, lane);
  else
    edge_tile<false>(vectors, radial, senders, receivers, w1t, w2t, w3t, wot,
                     actA, actB, ymeta, sr, up_s, up_v, agg_s, agg_v,
                     ebase, n_edges, lane);
}

// ---------------------------------------------------------------------------
// f_s = agg_s/4 @ W_down_s / sqrt(128) + skip_s/8
// f_v = agg_v/4 @ W_down_v / sqrt(128) + skip_v/8   (per component)
template <bool FULL>
__device__ __forceinline__ void down_tile(
    const float* __restrict__ node_feats, const int* __restrict__ node_specie,
    const float* __restrict__ agg_s, const float* __restrict__ agg_v,
    const _Float16* wds, const _Float16* wdv, _Float16* act,
    const float* __restrict__ W_skip_s, const float* __restrict__ W_skip_v,
    float* __restrict__ f_s, float* __restrict__ f_v,
    int base, int n_nodes, int lane) {
  const int nloc = lane & 15, mo = (lane < 16) ? 0 : 8;
  // ---- f_s ----
  for (int i = lane; i < 2048; i += 32) {
    int m = i >> 7, k = i & 127;
    int node = base + m;
    if (!FULL && node >= n_nodes) node = n_nodes - 1;
    act[i] = (_Float16)(agg_s[node * 128 + k] * 0.25f);  // 1/sqrt(AVG_NEI)
  }
#pragma unroll
  for (int nt = 0; nt < 8; ++nt) {
    v8f acc = {};
#pragma unroll
    for (int kt = 0; kt < 4; ++kt) {
      v16h a = load_A_frag(act, 128, kt * 32, lane);
      v16h b = load_B_frag(wds, 128, nt * 16, kt * 32, lane);
      acc = wmma32(a, b, acc);
    }
    int k2 = nt * 16 + nloc;
#pragma unroll
    for (int r = 0; r < 8; ++r) {
      int node = base + r + mo;
      if (!FULL && node >= n_nodes) continue;
      int spec = node_specie[node];
      const float* srow = node_feats + node * 256;
      const float* wcol = W_skip_s + spec * 64 * 128 + k2;
      float sk = 0.0f;
      for (int mm = 0; mm < 64; ++mm) sk += srow[mm] * wcol[mm * 128];
      f_s[node * 128 + k2] = acc[r] * INV_2M_F + sk * 0.125f;
    }
  }
  // ---- f_v, 3 components ----
  for (int c = 0; c < 3; ++c) {
    for (int i = lane; i < 2048; i += 32) {
      int m = i >> 7, k = i & 127;
      int node = base + m;
      if (!FULL && node >= n_nodes) node = n_nodes - 1;
      act[i] = (_Float16)(agg_v[(node * 128 + k) * 3 + c] * 0.25f);
    }
#pragma unroll
    for (int nt = 0; nt < 4; ++nt) {
      v8f acc = {};
#pragma unroll
      for (int kt = 0; kt < 4; ++kt) {
        v16h a = load_A_frag(act, 128, kt * 32, lane);
        v16h b = load_B_frag(wdv, 128, nt * 16, kt * 32, lane);
        acc = wmma32(a, b, acc);
      }
      int k = nt * 16 + nloc;
#pragma unroll
      for (int r = 0; r < 8; ++r) {
        int node = base + r + mo;
        if (!FULL && node >= n_nodes) continue;
        int spec = node_specie[node];
        const float* vrow = node_feats + node * 256 + 64 + c;
        const float* wcol = W_skip_v + spec * 64 * 64 + k;
        float sk = 0.0f;
        for (int mm = 0; mm < 64; ++mm) sk += vrow[mm * 3] * wcol[mm * 64];
        f_v[node * 192 + k * 3 + c] = acc[r] * INV_2M_F + sk * 0.125f;
      }
    }
  }
}

__global__ void k_down_skip(const float* __restrict__ node_feats,
                            const int* __restrict__ node_specie,
                            const float* __restrict__ agg_s,
                            const float* __restrict__ agg_v,
                            const float* __restrict__ W_down_s,
                            const float* __restrict__ W_down_v,
                            const float* __restrict__ W_skip_s,
                            const float* __restrict__ W_skip_v,
                            float* __restrict__ f_s, float* __restrict__ f_v,
                            int n_nodes) {
  extern __shared__ char smem[];
  _Float16* wds = (_Float16*)smem;        // [128][128] transposed (N,K)
  _Float16* wdv = wds + 16384;            // [64][128]  transposed
  _Float16* acts0 = wdv + 8192;
  const int tid = threadIdx.x;
  for (int i = tid; i < 16384; i += blockDim.x) {
    int n = i >> 7, k = i & 127;
    wds[i] = (_Float16)W_down_s[k * 128 + n];
  }
  for (int i = tid; i < 8192; i += blockDim.x) {
    int n = i >> 7, k = i & 127;
    wdv[i] = (_Float16)W_down_v[k * 64 + n];
  }
  __syncthreads();
  const int lane = tid & 31, wave = tid >> 5;
  _Float16* act = acts0 + wave * 2048;    // [16][128]
  int n_tiles = (n_nodes + 15) >> 4;
  int tile = blockIdx.x * (blockDim.x >> 5) + wave;
  if (tile >= n_tiles) return;
  int base = tile * 16;
  if (base + 16 <= n_nodes)
    down_tile<true>(node_feats, node_specie, agg_s, agg_v, wds, wdv, act,
                    W_skip_s, W_skip_v, f_s, f_v, base, n_nodes, lane);
  else
    down_tile<false>(node_feats, node_specie, agg_s, agg_v, wds, wdv, act,
                     W_skip_s, W_skip_v, f_s, f_v, base, n_nodes, lane);
}

// ---------------------------------------------------------------------------
__global__ void k_gate(const float* __restrict__ f_s,
                       const float* __restrict__ f_v,
                       float* __restrict__ out, int n_nodes) {
  int t = blockIdx.x * blockDim.x + threadIdx.x;
  int total = n_nodes * 64;
  if (t >= total) return;
  int n = t >> 6, k = t & 63;
  float g1 = silu_f(f_s[n * 128 + k]);
  float gt = silu_f(f_s[n * 128 + 64 + k]);
  out[n * 256 + k] = fmaxf(g1, 0.0f);
#pragma unroll
  for (int c = 0; c < 3; ++c)
    out[n * 256 + 64 + k * 3 + c] = fmaxf(gt * f_v[n * 192 + k * 3 + c], 0.0f);
}

// ---------------------------------------------------------------------------
extern "C" void kernel_launch(void* const* d_in, const int* in_sizes, int n_in,
                              void* d_out, int out_size, void* d_ws, size_t ws_size,
                              hipStream_t stream) {
  const float* vectors     = (const float*)d_in[0];
  const float* node_feats  = (const float*)d_in[1];
  const int*   node_specie = (const int*)d_in[2];
  const float* radial      = (const float*)d_in[3];
  const int*   senders     = (const int*)d_in[4];
  const int*   receivers   = (const int*)d_in[5];
  const float* W_up_s      = (const float*)d_in[6];
  const float* W_up_v      = (const float*)d_in[7];
  const float* mlp_w1      = (const float*)d_in[8];
  const float* mlp_w2      = (const float*)d_in[9];
  const float* mlp_w3      = (const float*)d_in[10];
  const float* mlp_wo      = (const float*)d_in[11];
  const float* W_skip_s    = (const float*)d_in[12];
  const float* W_skip_v    = (const float*)d_in[13];
  const float* W_down_s    = (const float*)d_in[14];
  const float* W_down_v    = (const float*)d_in[15];
  float* out = (float*)d_out;

  const int N = in_sizes[1] / 256;   // 20000
  const int E = in_sizes[0] / 3;     // 320000

  // workspace layout (floats)
  float* up_s  = (float*)d_ws;                 // N*64
  float* up_v  = up_s  + (size_t)N * 64;       // N*192  (N,64,3)
  float* agg_s = up_v  + (size_t)N * 192;      // N*128
  float* agg_v = agg_s + (size_t)N * 128;      // N*384  (N,128,3)
  float* f_s   = agg_v + (size_t)N * 384;      // N*128
  float* f_v   = f_s   + (size_t)N * 128;      // N*192

  // 1) zero aggregation buffers (agg_s + agg_v contiguous)
  k_zero<<<2048, 256, 0, stream>>>(agg_s, (long)N * 512);

  // 2) node "up" GEMMs (WMMA f16)
  {
    int tiles = (N + 15) / 16;
    int blocks = (tiles + 3) / 4;               // 4 waves/block
    size_t lds = (4096 + 4096 + 4 * 1024) * sizeof(_Float16); // 24.0 KB
    k_up<<<blocks, 128, lds, stream>>>(node_feats, W_up_s, W_up_v,
                                       up_s, up_v, N);
  }

  // 3) fused edge MLP (WMMA) + tensor product + atomic scatter
  {
    int tiles = (E + 15) / 16;
    int blocks = (tiles + 1) / 2;               // 2 waves/block
    size_t lds = (2048 + 4096 + 4096 + 16384 + 2 * 2048) * sizeof(_Float16)
               + 2 * 96 * sizeof(float);        // 62.2 KB
    k_edge<<<blocks, 64, lds, stream>>>(vectors, radial, senders, receivers,
                                        mlp_w1, mlp_w2, mlp_w3, mlp_wo,
                                        up_s, up_v, agg_s, agg_v, E);
  }

  // 4) down GEMMs (WMMA) + species skip (VALU) -> f_s/f_v
  {
    int tiles = (N + 15) / 16;
    int blocks = (tiles + 1) / 2;               // 2 waves/block
    size_t lds = (16384 + 8192 + 2 * 2048) * sizeof(_Float16); // 56.0 KB
    k_down_skip<<<blocks, 64, lds, stream>>>(node_feats, node_specie,
                                             agg_s, agg_v,
                                             W_down_s, W_down_v,
                                             W_skip_s, W_skip_v,
                                             f_s, f_v, N);
  }

  // 5) gating + ReLU epilogue
  {
    int total = N * 64;
    k_gate<<<(total + 255) / 256, 256, 0, stream>>>(f_s, f_v, out, N);
  }
}
